// Block_17179869258
// MI455X (gfx1250) — matrix operand order repaired and checked by reading, not compiled, not run on hardware
//
#include <hip/hip_runtime.h>
#include <hip/hip_fp16.h>
#include <cmath>

typedef _Float16 v16h __attribute__((ext_vector_type(16)));
typedef _Float16 h8   __attribute__((ext_vector_type(8)));
typedef float    v8f  __attribute__((ext_vector_type(8)));

#define DIMC   768
#define HEADS  12
#define HD     64
#define WIN    14
#define NTOK   196
#define NP     208          // token dim padded to 13*16
#define KPV    224          // PV K padded to 7*32
#define MLPH   3072
#define NWIN   50           // 2 batches * 5*5 windows
#define NBH    600          // NWIN * HEADS
#define MPAD   (NWIN*NP)    // 10400
#define PIX    8192         // 2*64*64
#define ASCALE 0.125f       // 64^-0.5

// ---------------- fragment helpers ----------------
__device__ __forceinline__ v16h ld_frag(const _Float16* p0, const _Float16* p1) {
  h8 lo = *reinterpret_cast<const h8*>(p0);
  h8 hi = *reinterpret_cast<const h8*>(p1);
  return __builtin_shufflevector(lo, hi, 0,1,2,3,4,5,6,7,8,9,10,11,12,13,14,15);
}
__device__ __forceinline__ v8f wmma32(v16h a, v16h b, v8f c) {
  return __builtin_amdgcn_wmma_f32_16x16x32_f16(false, a, false, b, (short)0, c, false, false);
}

// ---------------- CDNA5 async copy: global -> LDS (ASYNCcnt-tracked DMA) ----------------
__device__ __forceinline__ void async_g2l_b128(unsigned lds_addr, const _Float16* gaddr) {
  asm volatile("global_load_async_to_lds_b128 %0, %1, off"
               :: "v"(lds_addr), "v"(gaddr) : "memory");
}
__device__ __forceinline__ void wait_asynccnt0() {
  asm volatile("s_wait_asynccnt 0x0" ::: "memory");
}

// ---------------- weight transpose + f16 convert: W(KxN) -> Bt(NxK) ----------------
__global__ __launch_bounds__(256) void wt_kernel(const float* __restrict__ W,
                                                 _Float16* __restrict__ Bt,
                                                 int K, int N) {
  size_t i = (size_t)blockIdx.x * 256 + threadIdx.x;
  if (i >= (size_t)K * N) return;
  int k = (int)(i / N);
  int n = (int)(i % N);
  Bt[(size_t)n * K + k] = (_Float16)W[i];
}

// ---------------- LayerNorm (row of 768) ----------------
// mode 0: window-partitioned read from x (2,64,64,768), zero for padded tokens -> out (50,208,768) f16
// mode 1: direct row read (PIX,768) -> out f16
__global__ __launch_bounds__(256) void ln_kernel(const float* __restrict__ xin,
                                                 const float* __restrict__ g,
                                                 const float* __restrict__ b,
                                                 _Float16* __restrict__ out,
                                                 int mode) {
  __shared__ float tmp[256];
  int row = blockIdx.x;
  int tid = threadIdx.x;
  const float* src;
  if (mode == 0) {
    int wi = row / NP, t = row - wi * NP;
    bool valid = false; size_t off = 0;
    if (t < NTOK) {
      int bb = wi / 25, rem = wi - bb * 25;
      int wr = rem / 5, wc = rem - wr * 5;
      int hh = t / WIN, ww = t - hh * WIN;
      int H = wr * WIN + hh, W = wc * WIN + ww;
      if (H < 64 && W < 64) { valid = true; off = (((size_t)bb * 64 + H) * 64 + W) * DIMC; }
    }
    if (!valid) {
      for (int i = tid; i < DIMC; i += 256) out[(size_t)row * DIMC + i] = (_Float16)0.f;
      return;
    }
    src = xin + off;
  } else {
    src = xin + (size_t)row * DIMC;
  }
  float v0 = src[tid], v1 = src[tid + 256], v2 = src[tid + 512];
  tmp[tid] = v0 + v1 + v2;
  __syncthreads();
  for (int st = 128; st > 0; st >>= 1) { if (tid < st) tmp[tid] += tmp[tid + st]; __syncthreads(); }
  float mean = tmp[0] * (1.f / DIMC);
  __syncthreads();
  float d0 = v0 - mean, d1 = v1 - mean, d2 = v2 - mean;
  tmp[tid] = d0 * d0 + d1 * d1 + d2 * d2;
  __syncthreads();
  for (int st = 128; st > 0; st >>= 1) { if (tid < st) tmp[tid] += tmp[tid + st]; __syncthreads(); }
  float rinv = rsqrtf(tmp[0] * (1.f / DIMC) + 1e-6f);
  size_t base = (size_t)row * DIMC;
  out[base + tid      ] = (_Float16)(d0 * rinv * g[tid      ] + b[tid      ]);
  out[base + tid + 256] = (_Float16)(d1 * rinv * g[tid + 256] + b[tid + 256]);
  out[base + tid + 512] = (_Float16)(d2 * rinv * g[tid + 512] + b[tid + 512]);
}

// ---------------- WMMA GEMM, double-buffered async-LDS staged ----------------
// C(MxN) = A(MxK) * Bt(NxK)^T + bias. Block tile 128x128, K-step 32, 8 waves of 32x64.
// MODE 0: qkv split -> q (600,208,64) f16, k*ASCALE (600,208,64) f16, vT (600,64,224) f16
// MODE 1: proj -> window-unpartition + residual(res=x) -> of (2,64,64,768) f32
// MODE 2: fc1  -> exact GELU -> o0 f16 (PIX, N)
// MODE 3: fc2  -> + res -> of f32 (PIX, N)
template <int MODE>
__global__ __launch_bounds__(256) void gemm_kernel(const _Float16* __restrict__ A,
                                                   const _Float16* __restrict__ Bt,
                                                   const float* __restrict__ bias,
                                                   _Float16* __restrict__ o0,
                                                   _Float16* __restrict__ o1,
                                                   _Float16* __restrict__ o2,
                                                   float* __restrict__ of,
                                                   const float* __restrict__ res,
                                                   int M, int N, int K) {
  __shared__ __align__(16) _Float16 sA[2][128 * 32];
  __shared__ __align__(16) _Float16 sB[2][128 * 32];
  int tid = threadIdx.x;
  int w = tid >> 5, lane = tid & 31;
  int wm = w >> 1, wn = w & 1;
  int m0b = blockIdx.y * 128;
  int n0b = blockIdx.x * 128;
  int m0 = m0b + wm * 32;
  int n0 = n0b + wn * 64;
  int lRow = lane & 15, aKo = (lane >> 4) << 3, bKo = (lane >> 4) << 4;

  __builtin_prefetch(bias + n0b, 0, 0);

  unsigned sAaddr = (unsigned)(uintptr_t)&sA[0][0];
  unsigned sBaddr = (unsigned)(uintptr_t)&sB[0][0];

  // copy-index precompute: 512 16-byte chunks per 128x32 tile, 2 per thread
  int c0row = tid >> 2,          c0col = (tid & 3) << 3;          // chunk tid
  int c1row = (tid + 256) >> 2,  c1col = ((tid + 256) & 3) << 3;  // chunk tid+256
  int ra0 = min(m0b + c0row, M - 1), ra1 = min(m0b + c1row, M - 1);
  int rb0 = n0b + c0row,             rb1 = n0b + c1row;           // N is a multiple of 128
  const _Float16* gA0 = A  + (size_t)ra0 * K + c0col;
  const _Float16* gA1 = A  + (size_t)ra1 * K + c1col;
  const _Float16* gB0 = Bt + (size_t)rb0 * K + c0col;
  const _Float16* gB1 = Bt + (size_t)rb1 * K + c1col;
  unsigned lA0 = sAaddr + (unsigned)((c0row * 32 + c0col) * 2);
  unsigned lA1 = sAaddr + (unsigned)((c1row * 32 + c1col) * 2);
  unsigned lB0 = sBaddr + (unsigned)((c0row * 32 + c0col) * 2);
  unsigned lB1 = sBaddr + (unsigned)((c1row * 32 + c1col) * 2);

  auto issue_tile = [&](int buf, int k0) {
    unsigned boff = (unsigned)(buf * 128 * 32 * 2);
    async_g2l_b128(lA0 + boff, gA0 + k0);
    async_g2l_b128(lA1 + boff, gA1 + k0);
    async_g2l_b128(lB0 + boff, gB0 + k0);
    async_g2l_b128(lB1 + boff, gB1 + k0);
  };

  v8f acc[2][4] = {};
  int nK = K >> 5;
  issue_tile(0, 0);
  wait_asynccnt0();
  __syncthreads();
  for (int kt = 0; kt < nK; ++kt) {
    int cur = kt & 1;
    if (kt + 1 < nK) issue_tile(cur ^ 1, (kt + 1) << 5);
    const _Float16* aL0 = &sA[cur][(wm * 32 + lRow) * 32 + aKo];
    const _Float16* aL1 = aL0 + 16 * 32;
    v16h af0 = ld_frag(aL0, aL0 + 16);
    v16h af1 = ld_frag(aL1, aL1 + 16);
    #pragma unroll
    for (int j = 0; j < 4; ++j) {
      const _Float16* bL = &sB[cur][(wn * 64 + j * 16 + lRow) * 32 + aKo];
      v16h bf = ld_frag(bL, bL + 16);
      acc[0][j] = wmma32(af0, bf, acc[0][j]);
      acc[1][j] = wmma32(af1, bf, acc[1][j]);
    }
    wait_asynccnt0();
    __syncthreads();
  }

  // epilogue
  #pragma unroll
  for (int i = 0; i < 2; ++i) {
    int grb = m0 + i * 16 + ((lane >> 4) << 3);
    #pragma unroll
    for (int j = 0; j < 4; ++j) {
      int gc = n0 + j * 16 + (lane & 15);
      float bv = bias[gc];
      #pragma unroll
      for (int r = 0; r < 8; ++r) {
        int gr = grb + r;
        if (gr >= M) continue;
        float v = acc[i][j][r] + bv;
        if constexpr (MODE == 0) {
          int wi = gr / NP, t = gr - wi * NP;
          int sector = gc / DIMC;
          int c = gc - sector * DIMC;
          int hh = c >> 6, d = c & 63;
          size_t bh = (size_t)wi * HEADS + hh;
          if (sector == 0)      o0[(bh * NP + t) * HD + d] = (_Float16)v;            // q (unscaled)
          else if (sector == 1) o1[(bh * NP + t) * HD + d] = (_Float16)(v * ASCALE); // k * scale
          else                  o2[(bh * HD + d) * KPV + t] = (_Float16)v;           // v transposed
        } else if constexpr (MODE == 1) {
          int wi = gr / NP, t = gr - wi * NP;
          if (t < NTOK) {
            int bb = wi / 25, rem = wi - bb * 25;
            int wr = rem / 5, wc = rem - wr * 5;
            int hh = t / WIN, ww = t - hh * WIN;
            int H = wr * WIN + hh, W = wc * WIN + ww;
            if (H < 64 && W < 64) {
              size_t idx = (((size_t)bb * 64 + H) * 64 + W) * DIMC + gc;
              of[idx] = res[idx] + v;
            }
          }
        } else if constexpr (MODE == 2) {
          float gelu = 0.5f * v * (1.f + erff(v * 0.70710678118654752f));
          o0[(size_t)gr * N + gc] = (_Float16)gelu;
        } else {
          size_t idx = (size_t)gr * N + gc;
          of[idx] = res[idx] + v;
        }
      }
    }
  }
}

// ---------------- fused attention: scores + rel-pos + softmax + PV ----------------
// grid (13 strips, 600 window*head), block = 1 wave (32)
__global__ __launch_bounds__(32) void attn_kernel(const _Float16* __restrict__ q,
                                                  const _Float16* __restrict__ kbuf,
                                                  const _Float16* __restrict__ vt,
                                                  const float* __restrict__ relph,
                                                  const float* __restrict__ relpw,
                                                  _Float16* __restrict__ aw) {
  __shared__ __align__(16) float    sc[16][KPV];
  __shared__ __align__(16) _Float16 probs[16][KPV];
  __shared__ float relh[16][WIN];
  __shared__ float relw[16][WIN];
  int s = blockIdx.x;          // 16-row strip
  int bh = blockIdx.y;         // window*head
  int wi = bh / HEADS, hh = bh - wi * HEADS;
  int lane = threadIdx.x;
  const _Float16* qb = q    + (size_t)bh * NP * HD;
  const _Float16* kb = kbuf + (size_t)bh * NP * HD;
  const _Float16* vb = vt   + (size_t)bh * HD * KPV;

  // decomposed rel-pos tables for this strip: relh[m][k] = q_row . rel_pos_h[h - k + 13]
  for (int idx = lane; idx < 2 * 16 * WIN; idx += 32) {
    int table = idx >= 16 * WIN;
    int e = idx - table * 16 * WIN;
    int m = e / WIN, kk = e - (e / WIN) * WIN;
    int r = s * 16 + m;
    float a = 0.f;
    if (r < NTOK) {
      int pos = table ? (r % WIN) : (r / WIN);
      const float* rp = (table ? relpw : relph) + (size_t)(pos - kk + WIN - 1) * HD;
      const _Float16* qr = qb + (size_t)r * HD;
      #pragma unroll 8
      for (int c = 0; c < HD; ++c) a += (float)qr[c] * rp[c];
    }
    if (table) relw[m][kk] = a; else relh[m][kk] = a;
  }
  __syncthreads();

  int lRow = lane & 15, aKo = (lane >> 4) << 3, bKo = (lane >> 4) << 4;
  // A fragments: q strip rows, K = 64 -> 2 k-steps
  const _Float16* qa = qb + (size_t)(s * 16 + lRow) * HD + aKo;
  v16h aF0 = ld_frag(qa,      qa + 16);
  v16h aF1 = ld_frag(qa + 32, qa + 48);
  // scores: 13 column tiles
  v8f acc[13];
  #pragma unroll
  for (int nt = 0; nt < 13; ++nt) {
    acc[nt] = v8f{};
    const _Float16* kp = kb + (size_t)(nt * 16 + lRow) * HD + aKo;
    v16h bF0 = ld_frag(kp,      kp + 16);
    v16h bF1 = ld_frag(kp + 32, kp + 48);
    acc[nt] = wmma32(aF0, bF0, acc[nt]);
    acc[nt] = wmma32(aF1, bF1, acc[nt]);
  }
  // add rel-pos, mask fake padding, stash in LDS
  #pragma unroll
  for (int nt = 0; nt < 13; ++nt) {
    int col = nt * 16 + (lane & 15);
    int k2 = col / WIN, w2 = col - k2 * WIN;
    #pragma unroll
    for (int r = 0; r < 8; ++r) {
      int m = ((lane >> 4) << 3) + r;
      float v = acc[nt][r];
      v = (col < NTOK) ? (v + relh[m][k2] + relw[m][w2]) : -1e30f;
      sc[m][col] = v;
    }
  }
  __syncthreads();
  // row softmax (lanes 0..15 own one row each), write f16 probs padded to 224
  if (lane < 16) {
    int m = lane;
    float mx = -1e30f;
    for (int c = 0; c < NP; ++c) mx = fmaxf(mx, sc[m][c]);
    float sum = 0.f;
    for (int c = 0; c < NP; ++c) { float e = expf(sc[m][c] - mx); sc[m][c] = e; sum += e; }
    float inv = 1.f / sum;
    for (int c = 0; c < KPV; ++c) probs[m][c] = (_Float16)(c < NP ? sc[m][c] * inv : 0.f);
  }
  __syncthreads();
  // PV: out(16 x 64) = probs(16 x 224) * v(224 x 64); vT rows are zero-padded past 208
  v8f acc2[4] = {};
  #pragma unroll
  for (int ks = 0; ks < 7; ++ks) {
    const _Float16* pa = &probs[lRow][ks * 32 + aKo];
    v16h aP = ld_frag(pa, pa + 16);
    #pragma unroll
    for (int j = 0; j < 4; ++j) {
      const _Float16* pb = vb + (size_t)(j * 16 + lRow) * KPV + ks * 32 + aKo;
      v16h bF = ld_frag(pb, pb + 16);
      acc2[j] = wmma32(aP, bF, acc2[j]);
    }
  }
  // write head-concatenated output (50,208,768) f16
  #pragma unroll
  for (int j = 0; j < 4; ++j) {
    int d = j * 16 + (lane & 15);
    #pragma unroll
    for (int r = 0; r < 8; ++r) {
      int m = ((lane >> 4) << 3) + r;
      aw[((size_t)wi * NP + s * 16 + m) * DIMC + hh * HD + d] = (_Float16)acc2[j][r];
    }
  }
}

// ---------------- host-side orchestration ----------------
static inline size_t alignup(size_t x) { return (x + 255) & ~(size_t)255; }

extern "C" void kernel_launch(void* const* d_in, const int* in_sizes, int n_in,
                              void* d_out, int out_size, void* d_ws, size_t ws_size,
                              hipStream_t stream) {
  (void)in_sizes; (void)n_in; (void)out_size; (void)ws_size;
  const float* x      = (const float*)d_in[0];
  const float* ln1_g  = (const float*)d_in[1];
  const float* ln1_b  = (const float*)d_in[2];
  const float* qkv_w  = (const float*)d_in[3];
  const float* qkv_b  = (const float*)d_in[4];
  const float* proj_w = (const float*)d_in[5];
  const float* proj_b = (const float*)d_in[6];
  const float* rel_h  = (const float*)d_in[7];
  const float* rel_w  = (const float*)d_in[8];
  const float* ln2_g  = (const float*)d_in[9];
  const float* ln2_b  = (const float*)d_in[10];
  const float* fc1_w  = (const float*)d_in[11];
  const float* fc1_b  = (const float*)d_in[12];
  const float* fc2_w  = (const float*)d_in[13];
  const float* fc2_b  = (const float*)d_in[14];
  float* out = (float*)d_out;

  char* ws = (char*)d_ws;
  size_t off = 0;
  auto take = [&](size_t bytes) { char* p = ws + off; off += alignup(bytes); return p; };
  _Float16* xw    = (_Float16*)take((size_t)MPAD * DIMC * 2);          // LN1, windowed, padded
  _Float16* qbuf  = (_Float16*)take((size_t)NBH * NP * HD * 2);
  _Float16* kbuf2 = (_Float16*)take((size_t)NBH * NP * HD * 2);
  size_t vt_bytes = (size_t)NBH * HD * KPV * 2;
  _Float16* vtbuf = (_Float16*)take(vt_bytes);
  _Float16* awbuf = (_Float16*)take((size_t)MPAD * DIMC * 2);
  float*    x1    = (float*)   take((size_t)PIX * DIMC * 4);
  _Float16* h2    = (_Float16*)take((size_t)PIX * DIMC * 2);
  _Float16* hid   = (_Float16*)take((size_t)PIX * MLPH * 2);
  _Float16* btqkv = (_Float16*)take((size_t)3 * DIMC * DIMC * 2);
  _Float16* btprj = (_Float16*)take((size_t)DIMC * DIMC * 2);
  _Float16* btfc1 = (_Float16*)take((size_t)DIMC * MLPH * 2);
  _Float16* btfc2 = (_Float16*)take((size_t)MLPH * DIMC * 2);

  auto wtgrid = [](size_t n) { return dim3((unsigned)((n + 255) / 256)); };
  wt_kernel<<<wtgrid((size_t)DIMC * 3 * DIMC), 256, 0, stream>>>(qkv_w, btqkv, DIMC, 3 * DIMC);
  wt_kernel<<<wtgrid((size_t)DIMC * DIMC),     256, 0, stream>>>(proj_w, btprj, DIMC, DIMC);
  wt_kernel<<<wtgrid((size_t)DIMC * MLPH),     256, 0, stream>>>(fc1_w, btfc1, DIMC, MLPH);
  wt_kernel<<<wtgrid((size_t)MLPH * DIMC),     256, 0, stream>>>(fc2_w, btfc2, MLPH, DIMC);

  ln_kernel<<<dim3(MPAD), 256, 0, stream>>>(x, ln1_g, ln1_b, xw, 0);

  hipMemsetAsync(vtbuf, 0, vt_bytes, stream);  // zero the K-pad rows of v^T

  gemm_kernel<0><<<dim3(3 * DIMC / 128, (MPAD + 127) / 128), 256, 0, stream>>>(
      xw, btqkv, qkv_b, qbuf, kbuf2, vtbuf, nullptr, nullptr, MPAD, 3 * DIMC, DIMC);

  attn_kernel<<<dim3(NP / 16, NBH), 32, 0, stream>>>(qbuf, kbuf2, vtbuf, rel_h, rel_w, awbuf);

  gemm_kernel<1><<<dim3(DIMC / 128, (MPAD + 127) / 128), 256, 0, stream>>>(
      awbuf, btprj, proj_b, nullptr, nullptr, nullptr, x1, x, MPAD, DIMC, DIMC);

  ln_kernel<<<dim3(PIX), 256, 0, stream>>>(x1, ln2_g, ln2_b, h2, 1);

  gemm_kernel<2><<<dim3(MLPH / 128, PIX / 128), 256, 0, stream>>>(
      h2, btfc1, fc1_b, hid, nullptr, nullptr, nullptr, nullptr, PIX, MLPH, DIMC);

  gemm_kernel<3><<<dim3(DIMC / 128, PIX / 128), 256, 0, stream>>>(
      hid, btfc2, fc2_b, nullptr, nullptr, nullptr, out, x1, PIX, DIMC, MLPH);
}
